// SwinTransformerBlock3D_73280732004618
// MI455X (gfx1250) — matrix operand order repaired and checked by reading, not compiled, not run on hardware
//
#include <hip/hip_runtime.h>
#include <hip/hip_bf16.h>
#include <math.h>

typedef __attribute__((ext_vector_type(16))) _Float16 v16h;
typedef __attribute__((ext_vector_type(8)))  _Float16 v8h;
typedef __attribute__((ext_vector_type(8)))  float    v8f;

#define C_DIM   192
#define HEADS   6
#define HD      32
#define NTOK    98
#define NPAD    112
#define NWIN    1520            // 8 * 10 * 19 windows on 16x70x133 padded grid
#define MWIN    (NWIN * NTOK)   // 148960 window tokens
#define MTOK    (16 * 64 * 128) // 131072 real tokens
#define QKV_SEC ((long)NWIN * HEADS * NTOK * HD) // 28,600,320 elems per q/k/v

// ---------------------------------------------------------------------------
// WMMA helpers (CDNA5 layouts, wave32)
// A 16x32 f16 tile, row-major source with leading dim `ld` (f16 elems):
//   lanes 0-15 hold row (lane), K = {0..7, 16..23};  lanes 16-31: K = {8..15, 24..31}
__device__ __forceinline__ v16h ld_afrag(const _Float16* base, int ld) {
  int lane = threadIdx.x & 31;
  const _Float16* p = base + (lane & 15) * ld + ((lane & 16) ? 8 : 0);
  v8h lo = *(const v8h*)p;
  v8h hi = *(const v8h*)(p + 16);
  v16h r;
#pragma unroll
  for (int i = 0; i < 8; ++i) { r[i] = lo[i]; r[i + 8] = hi[i]; }
  return r;
}
// B 32x16 f16 tile from an N-major (transposed) staging buffer with leading dim ld:
//   lanes 0-15 hold col (lane) K = {0..7, 16..23};  lanes 16-31: K = {8..15, 24..31}
__device__ __forceinline__ v16h ld_bfrag(const _Float16* baseT, int ld) {
  int lane = threadIdx.x & 31;
  const _Float16* p = baseT + (lane & 15) * ld + ((lane & 16) ? 8 : 0);
  v8h lo = *(const v8h*)p;
  v8h hi = *(const v8h*)(p + 16);
  v16h r;
#pragma unroll
  for (int i = 0; i < 8; ++i) { r[i] = lo[i]; r[i + 8] = hi[i]; }
  return r;
}
__device__ __forceinline__ v8f wmma16x16x32(v16h a, v16h b, v8f c) {
  return __builtin_amdgcn_wmma_f32_16x16x32_f16(false, a, false, b, (short)0, c,
                                                false, false);
}
__device__ __forceinline__ float wred_sum32(float v) {
#pragma unroll
  for (int m = 1; m < 32; m <<= 1) v += __shfl_xor(v, m, 32);
  return v;
}
__device__ __forceinline__ float hred_max16(float v) {
#pragma unroll
  for (int m = 1; m < 16; m <<= 1) v = fmaxf(v, __shfl_xor(v, m, 32));
  return v;
}
__device__ __forceinline__ float hred_sum16(float v) {
#pragma unroll
  for (int m = 1; m < 16; m <<= 1) v += __shfl_xor(v, m, 32);
  return v;
}

// ---------------------------------------------------------------------------
// K0: mod = silu(t) @ wa1 @ wa2 + ba2   (1x192 @ 192x24 @ 24x1152)
__global__ void mod_kernel(const float* __restrict__ t, const float* __restrict__ wa1,
                           const float* __restrict__ wa2, const float* __restrict__ ba2,
                           float* __restrict__ mod) {
  __shared__ float s[192];
  __shared__ float a[24];
  int tid = threadIdx.x;
  if (tid < 192) { float v = t[tid]; s[tid] = v / (1.f + expf(-v)); }
  __syncthreads();
  if (tid < 24) {
    float acc = 0.f;
    for (int i = 0; i < 192; ++i) acc += s[i] * wa1[i * 24 + tid];
    a[tid] = acc;
  }
  __syncthreads();
  for (int j = tid; j < 1152; j += 256) {
    float acc = ba2[j];
#pragma unroll
    for (int r = 0; r < 24; ++r) acc += a[r] * wa2[r * 1152 + j];
    mod[j] = acc;
  }
}

// Kw: weight f32 -> f16
__global__ void cvt_kernel(const float* __restrict__ s, _Float16* __restrict__ d, int n) {
  int i = blockIdx.x * 256 + threadIdx.x;
  if (i < n) d[i] = (_Float16)s[i];
}

// ---------------------------------------------------------------------------
// K1: LN + modulate + pad/roll/window-partition gather -> xw f16 (MWIN x 192)
// one wave per window-token; lane c holds channels {c, c+32, ..., c+160}
__global__ void __launch_bounds__(256) lnwin_kernel(const float* __restrict__ x,
                                                    const float* __restrict__ mod,
                                                    _Float16* __restrict__ xw) {
  int wave = threadIdx.x >> 5, lane = threadIdx.x & 31;
  long tw = (long)blockIdx.x * 8 + wave;
  if (tw >= MWIN) return;
  int win = (int)(tw / NTOK), tok = (int)(tw % NTOK);
  int ww = win % 19, wh = (win / 19) % 10, wd = win / 190;
  int w0 = tok % 7, h0 = (tok / 7) % 7, d0 = tok / 49;
  int d = (wd * 2 + d0 + 1) % 16;        // undo roll(-1) on padded depth 16
  int h = (wh * 7 + h0 + 3) % 70;        // undo roll(-3) on padded height 70
  int w = (ww * 7 + w0 + 3) % 133;       // undo roll(-3) on padded width 133
  bool valid = (h < 64) && (w < 128);
  _Float16* o = xw + tw * C_DIM;
  if (!valid) {
#pragma unroll
    for (int c = 0; c < 6; ++c) o[lane + 32 * c] = (_Float16)0.f;
    return;
  }
  const float* xp = x + ((long)(d * 64 + h) * 128 + w) * C_DIM;
  float v[6], sum = 0.f;
#pragma unroll
  for (int c = 0; c < 6; ++c) { v[c] = xp[lane + 32 * c]; sum += v[c]; }
  float mean = wred_sum32(sum) * (1.f / 192.f);
  float var = 0.f;
#pragma unroll
  for (int c = 0; c < 6; ++c) { float dv = v[c] - mean; var += dv * dv; }
  var = wred_sum32(var) * (1.f / 192.f);
  float rs = rsqrtf(var + 1e-6f);
#pragma unroll
  for (int c = 0; c < 6; ++c) {
    int ch = lane + 32 * c;
    float y = (v[c] - mean) * rs * (1.f + mod[192 + ch]) + mod[ch];
    o[ch] = (_Float16)y;
  }
}

// K5: LN + modulate for MLP input (reads x1 from d_out)
__global__ void __launch_bounds__(256) lnmlp_kernel(const float* __restrict__ x1,
                                                    const float* __restrict__ mod,
                                                    _Float16* __restrict__ y2) {
  int wave = threadIdx.x >> 5, lane = threadIdx.x & 31;
  long tk = (long)blockIdx.x * 8 + wave;
  if (tk >= MTOK) return;
  const float* xp = x1 + tk * C_DIM;
  float v[6], sum = 0.f;
#pragma unroll
  for (int c = 0; c < 6; ++c) { v[c] = xp[lane + 32 * c]; sum += v[c]; }
  float mean = wred_sum32(sum) * (1.f / 192.f);
  float var = 0.f;
#pragma unroll
  for (int c = 0; c < 6; ++c) { float dv = v[c] - mean; var += dv * dv; }
  var = wred_sum32(var) * (1.f / 192.f);
  float rs = rsqrtf(var + 1e-6f);
  _Float16* o = y2 + tk * C_DIM;
#pragma unroll
  for (int c = 0; c < 6; ++c) {
    int ch = lane + 32 * c;
    float y = (v[c] - mean) * rs * (1.f + mod[768 + ch]) + mod[576 + ch];
    o[ch] = (_Float16)y;
  }
}

// ---------------------------------------------------------------------------
// Generic f16 WMMA GEMM: 128(M) x 64(N) block tile, 256 threads = 8 waves (4x2),
// each wave owns a 32x32 tile (2x2 WMMA 16x16x32 accumulators).
// K staged 64 deep per LDS round (8 WMMAs per barrier pair) with global_prefetch
// of the next K-slab overlapping the compute clause.
// EPI: 0=qkv-scatter(+scale q), 1=proj + window-reverse + residual, 2=GELU->f16,
//      3=final x1 + g_mlp*h -> f32 out
template <int EPI>
__global__ void __launch_bounds__(256) gemm_kernel(
    const _Float16* __restrict__ A, const _Float16* __restrict__ B, int M, int N,
    int K, const float* __restrict__ bias, const float* __restrict__ gmod,
    const float* __restrict__ resid, float* __restrict__ outf,
    _Float16* __restrict__ outh) {
  __shared__ __align__(16) _Float16 As[128 * 64];  // 16 KB
  __shared__ __align__(16) _Float16 Bt[64 * 64];   // 8 KB, N-major transposed
  int nblk = N >> 6;
  int bm = (blockIdx.x / nblk) * 128;
  int bn = (blockIdx.x % nblk) * 64;
  int tid = threadIdx.x, wave = tid >> 5, lane = tid & 31;
  int wm = (wave >> 1) * 32, wn = (wave & 1) * 32;
  v8f acc[2][2] = {};
  for (int k0 = 0; k0 < K; k0 += 64) {
    {  // stage A: 128 rows x 64 cols; each thread one 32-elem half row
      int row = tid >> 1, half = tid & 1;
      long gm = bm + row;
      v8h a0, a1, a2, a3;
      if (gm < M) {
        const _Float16* src = A + gm * (long)K + k0 + half * 32;
        a0 = *(const v8h*)src;
        a1 = *(const v8h*)(src + 8);
        a2 = *(const v8h*)(src + 16);
        a3 = *(const v8h*)(src + 24);
        if (k0 + 64 < K) __builtin_prefetch(src + 64, 0, 3);  // next K-slab
      } else {
#pragma unroll
        for (int i = 0; i < 8; ++i) {
          a0[i] = (_Float16)0.f; a1[i] = (_Float16)0.f;
          a2[i] = (_Float16)0.f; a3[i] = (_Float16)0.f;
        }
      }
      _Float16* dst = As + row * 64 + half * 32;
      *(v8h*)(dst) = a0;
      *(v8h*)(dst + 8) = a1;
      *(v8h*)(dst + 16) = a2;
      *(v8h*)(dst + 24) = a3;
    }
    {  // stage B transposed: 64 K-rows x 64 cols -> Bt[n][k]
      int kk = tid >> 2, ng = tid & 3;
      const _Float16* src = B + (long)(k0 + kk) * N + bn + ng * 16;
      v8h b0 = *(const v8h*)src;
      v8h b1 = *(const v8h*)(src + 8);
      if (k0 + 64 < K) __builtin_prefetch(src + (long)64 * N, 0, 3);
#pragma unroll
      for (int i = 0; i < 8; ++i) {
        Bt[(ng * 16 + i) * 64 + kk] = b0[i];
        Bt[(ng * 16 + 8 + i) * 64 + kk] = b1[i];
      }
    }
    __syncthreads();
#pragma unroll
    for (int kk = 0; kk < 2; ++kk)
#pragma unroll
      for (int i = 0; i < 2; ++i) {
        v16h af = ld_afrag(As + (wm + 16 * i) * 64 + kk * 32, 64);
#pragma unroll
        for (int j = 0; j < 2; ++j) {
          v16h bf = ld_bfrag(Bt + (wn + 16 * j) * 64 + kk * 32, 64);
          acc[i][j] = wmma16x16x32(af, bf, acc[i][j]);
        }
      }
    __syncthreads();
  }
  // epilogue: C/D layout -> element (r,lane): M = r + (lane&16 ? 8:0), N = lane&15
#pragma unroll
  for (int i = 0; i < 2; ++i)
#pragma unroll
    for (int j = 0; j < 2; ++j)
#pragma unroll
      for (int r = 0; r < 8; ++r) {
        int m = bm + wm + 16 * i + r + ((lane & 16) ? 8 : 0);
        int n = bn + wn + 16 * j + (lane & 15);
        if (m >= M) continue;
        float val = acc[i][j][r] + bias[n];
        if (EPI == 0) {  // qkv: scatter to q/k/v per-(window,head) layout
          int mat = n / 192, hidx = (n % 192) / 32, dch = n % 32;
          if (mat == 0) val *= 0.17677669529663687f;  // hd^-0.5
          int win = m / NTOK, tok = m % NTOK;
          long off = (long)mat * QKV_SEC +
                     (((long)(win * HEADS + hidx) * NTOK + tok) * HD + dch);
          outh[off] = (_Float16)val;
        } else if (EPI == 1) {  // proj: window-reverse + roll-back + crop + residual
          int win = m / NTOK, tok = m % NTOK;
          int ww = win % 19, wh = (win / 19) % 10, wd = win / 190;
          int w0 = tok % 7, h0 = (tok / 7) % 7, d0 = tok / 49;
          int d = (wd * 2 + d0 + 1) % 16;
          int h = (wh * 7 + h0 + 3) % 70;
          int w = (ww * 7 + w0 + 3) % 133;
          if (h < 64 && w < 128) {
            long o = ((long)(d * 64 + h) * 128 + w) * C_DIM + n;
            outf[o] = resid[o] + gmod[n] * val;
          }
        } else if (EPI == 2) {  // exact GELU -> f16 hidden
          float g = 0.5f * val * (1.f + erff(val * 0.7071067811865475f));
          outh[(long)m * N + n] = (_Float16)g;
        } else {  // final: x1 + g_mlp * h
          long o = (long)m * C_DIM + n;
          outf[o] = resid[o] + gmod[n] * val;
        }
      }
}

// ---------------------------------------------------------------------------
// K3: attention, one block (256 thr, 8 waves) per (window, head).
// Q,K,Vt staged in LDS (112-padded); wave w owns score rows [16w,16w+16).
// S = Q K^T (WMMA, K=32), + rel-pos bias (index computed on the fly) + mask,
// softmax with shfl row reductions, P -> LDS, O = P V (WMMA, K=128 padded).
__global__ void __launch_bounds__(256) attn_kernel(
    const _Float16* __restrict__ qb, const _Float16* __restrict__ kb,
    const _Float16* __restrict__ vb, const float* __restrict__ rel_bias,
    const float* __restrict__ maskm, _Float16* __restrict__ outw) {
  __shared__ __align__(16) _Float16 smem[3 * NPAD * HD + NPAD * 128];
  _Float16* q_s = smem;                  // 112 x 32
  _Float16* k_s = smem + NPAD * HD;      // 112 x 32  (row-major == Bt for QK^T)
  _Float16* vT  = smem + 2 * NPAD * HD;  // 32 x 112  (N-major Bt for P·V)
  _Float16* P   = smem + 3 * NPAD * HD;  // 112 x 128
  int tid = threadIdx.x, wave = tid >> 5, lane = tid & 31;
  int win = blockIdx.x / HEADS, head = blockIdx.x % HEADS;
  // zero all staging (pads must be 0)
  {
    unsigned* z = (unsigned*)smem;
    int n32 = (3 * NPAD * HD + NPAD * 128) / 2;
    for (int i = tid; i < n32; i += 256) z[i] = 0u;
  }
  __syncthreads();
  long base = (long)(win * HEADS + head) * NTOK * HD;
  for (int e = tid; e < NTOK * 4; e += 256) {  // 98 rows x 4 v8h chunks
    int row = e >> 2, c = (e & 3) * 8;
    *(v8h*)(q_s + row * HD + c) = *(const v8h*)(qb + base + row * HD + c);
    *(v8h*)(k_s + row * HD + c) = *(const v8h*)(kb + base + row * HD + c);
  }
  for (int e = tid; e < NTOK * HD; e += 256) {  // transpose V
    int r = e >> 5, c = e & 31;
    vT[c * NPAD + r] = vb[base + e];
  }
  __syncthreads();

  int m0 = wave * 16;
  if (wave < 7) {
    v8f s[7];
    v16h aq = ld_afrag(q_s + m0 * HD, HD);
#pragma unroll
    for (int tn = 0; tn < 7; ++tn) {
      v8f z = {};
      v16h bk = ld_bfrag(k_s + tn * 16 * HD, HD);
      s[tn] = wmma16x16x32(aq, bk, z);
    }
    float sc[7][8];
#pragma unroll
    for (int tn = 0; tn < 7; ++tn)
#pragma unroll
      for (int r = 0; r < 8; ++r) {
        int m = m0 + r + ((lane & 16) ? 8 : 0);
        int n = tn * 16 + (lane & 15);
        float v = s[tn][r];
        if (m < NTOK && n < NTOK) {
          int dm = m / 49, hm = (m / 7) % 7, wm = m % 7;
          int dn = n / 49, hn = (n / 7) % 7, wn = n % 7;
          int idx = (dm - dn + 1) * 169 + (hm - hn + 6) * 13 + (wm - wn + 6);
          v += rel_bias[idx * HEADS + head] +
               maskm[((long)win * NTOK + m) * NTOK + n];
        } else {
          v = -1e30f;
        }
        sc[tn][r] = v;
      }
#pragma unroll
    for (int r = 0; r < 8; ++r) {
      float mx = -1e30f;
#pragma unroll
      for (int tn = 0; tn < 7; ++tn) mx = fmaxf(mx, sc[tn][r]);
      mx = hred_max16(mx);
      float sum = 0.f;
#pragma unroll
      for (int tn = 0; tn < 7; ++tn) {
        float e = __expf(sc[tn][r] - mx);
        sc[tn][r] = e;
        sum += e;
      }
      sum = hred_sum16(sum);
      float inv = 1.f / sum;
#pragma unroll
      for (int tn = 0; tn < 7; ++tn) sc[tn][r] *= inv;
    }
#pragma unroll
    for (int tn = 0; tn < 7; ++tn)
#pragma unroll
      for (int r = 0; r < 8; ++r) {
        int m = m0 + r + ((lane & 16) ? 8 : 0);
        P[m * 128 + tn * 16 + (lane & 15)] = (_Float16)sc[tn][r];
      }
  }
  __syncthreads();
  if (wave < 7) {
    v8f o0 = {}, o1 = {};
#pragma unroll
    for (int kc = 0; kc < 4; ++kc) {
      v16h ap = ld_afrag(P + m0 * 128 + kc * 32, 128);
      v16h b0 = ld_bfrag(vT + kc * 32, NPAD);
      v16h b1 = ld_bfrag(vT + 16 * NPAD + kc * 32, NPAD);
      o0 = wmma16x16x32(ap, b0, o0);
      o1 = wmma16x16x32(ap, b1, o1);
    }
#pragma unroll
    for (int r = 0; r < 8; ++r) {
      int m = m0 + r + ((lane & 16) ? 8 : 0);
      if (m < NTOK) {
        long o = ((long)win * NTOK + m) * C_DIM + head * HD;
        outw[o + (lane & 15)] = (_Float16)o0[r];
        outw[o + 16 + (lane & 15)] = (_Float16)o1[r];
      }
    }
  }
}

// ---------------------------------------------------------------------------
extern "C" void kernel_launch(void* const* d_in, const int* in_sizes, int n_in,
                              void* d_out, int out_size, void* d_ws, size_t ws_size,
                              hipStream_t stream) {
  (void)in_sizes; (void)n_in; (void)out_size; (void)ws_size;
  const float* x        = (const float*)d_in[0];
  const float* maskm    = (const float*)d_in[1];
  const float* tvec     = (const float*)d_in[2];
  const float* w_qkv    = (const float*)d_in[3];
  const float* b_qkv    = (const float*)d_in[4];
  const float* rel_bias = (const float*)d_in[5];
  const float* w_proj   = (const float*)d_in[6];
  const float* b_proj   = (const float*)d_in[7];
  const float* w1       = (const float*)d_in[8];
  const float* b1       = (const float*)d_in[9];
  const float* w2       = (const float*)d_in[10];
  const float* b2       = (const float*)d_in[11];
  const float* wa1      = (const float*)d_in[12];
  const float* wa2      = (const float*)d_in[13];
  const float* ba2      = (const float*)d_in[14];
  float* out = (float*)d_out;

  char* ws = (char*)d_ws;
  size_t off = 0;
  auto alloc = [&](size_t bytes) -> char* {
    char* p = ws + off;
    off = (off + bytes + 255) & ~(size_t)255;
    return p;
  };
  float*    modb   = (float*)alloc(1152 * sizeof(float));
  _Float16* wqkvh  = (_Float16*)alloc((size_t)192 * 576 * 2);
  _Float16* wprojh = (_Float16*)alloc((size_t)192 * 192 * 2);
  _Float16* w1h    = (_Float16*)alloc((size_t)192 * 768 * 2);
  _Float16* w2h    = (_Float16*)alloc((size_t)768 * 192 * 2);
  _Float16* xw     = (_Float16*)alloc((size_t)MWIN * C_DIM * 2);
  _Float16* qkvb   = (_Float16*)alloc((size_t)3 * QKV_SEC * 2);
  _Float16* hidden = (_Float16*)alloc((size_t)MTOK * 768 * 2);
  _Float16* outw = xw;    // alias: xw dead after QKV GEMM
  _Float16* y2   = qkvb;  // alias: q/k/v dead after attention

  mod_kernel<<<1, 256, 0, stream>>>(tvec, wa1, wa2, ba2, modb);
  cvt_kernel<<<(192 * 576 + 255) / 256, 256, 0, stream>>>(w_qkv, wqkvh, 192 * 576);
  cvt_kernel<<<(192 * 192 + 255) / 256, 256, 0, stream>>>(w_proj, wprojh, 192 * 192);
  cvt_kernel<<<(192 * 768 + 255) / 256, 256, 0, stream>>>(w1, w1h, 192 * 768);
  cvt_kernel<<<(768 * 192 + 255) / 256, 256, 0, stream>>>(w2, w2h, 768 * 192);

  lnwin_kernel<<<(MWIN + 7) / 8, 256, 0, stream>>>(x, modb, xw);

  int mtw = (MWIN + 127) / 128;  // 1164
  gemm_kernel<0><<<mtw * (576 / 64), 256, 0, stream>>>(
      xw, wqkvh, MWIN, 576, 192, b_qkv, nullptr, nullptr, nullptr, qkvb);

  attn_kernel<<<NWIN * HEADS, 256, 0, stream>>>(
      qkvb, qkvb + QKV_SEC, qkvb + 2 * QKV_SEC, rel_bias, maskm, outw);

  gemm_kernel<1><<<mtw * (192 / 64), 256, 0, stream>>>(
      outw, wprojh, MWIN, 192, 192, b_proj, modb + 384, x, out, nullptr);

  lnmlp_kernel<<<MTOK / 8, 256, 0, stream>>>(out, modb, y2);

  int mtt = MTOK / 128;  // 1024
  gemm_kernel<2><<<mtt * (768 / 64), 256, 0, stream>>>(
      y2, w1h, MTOK, 768, 192, b1, nullptr, nullptr, nullptr, hidden);
  gemm_kernel<3><<<mtt * (192 / 64), 256, 0, stream>>>(
      hidden, w2h, MTOK, 192, 768, b2, modb + 960, out, out, nullptr);
}
